// SpikeEncoder_11003706212829
// MI455X (gfx1250) — compile-verified, matrix operand order checked by reading, not run on hardware
//
#include <hip/hip_runtime.h>
#include <hip/hip_bf16.h>
#include <stdint.h>

// Problem constants (match reference)
#define N_NEURONS   512
#define SEQ_LEN     2048
#define NEVENTS     65536
#define BVAL        16
#define ROWS        (BVAL * N_NEURONS)          // 8192 segment rows
#define MAXE        64                          // max events/row (Poisson mean=8)
#define N_TILES     (SEQ_LEN / 16)              // 128 time tiles of 16
#define OUT_ELEMS   (BVAL * N_NEURONS * SEQ_LEN)

// g(d) = COEF * exp(-0.125*d^2) = COEF * exp2(QSCALE*d^2)
#define QSCALE  (-0.18033688011112042f)   // -0.125 * log2(e)
#define COEF    (0.19947114020071635f)    // 1/(sigma*sqrt(2*pi)), sigma=2

typedef __attribute__((ext_vector_type(16))) _Float16 v16h;
typedef __attribute__((ext_vector_type(8)))  float    v8f;

__device__ __forceinline__ float fast_exp2(float x) {
#if __has_builtin(__builtin_amdgcn_exp2f)
  return __builtin_amdgcn_exp2f(x);     // v_exp_f32; underflows cleanly to 0
#else
  return exp2f(x);
#endif
}

// ---------------- kernel 0: zero output + per-row counters ----------------
__global__ __launch_bounds__(256) void k_zero(float4* __restrict__ out4,
                                              int* __restrict__ counts) {
  int i = blockIdx.x * blockDim.x + threadIdx.x;
  if (i < OUT_ELEMS / 4) out4[i] = make_float4(0.f, 0.f, 0.f, 0.f);
  if (i < ROWS) counts[i] = 0;
}

// ---------------- kernel 1: bin events by (batch, neuron) row -------------
__global__ __launch_bounds__(256) void k_bin(const float* __restrict__ events,
                                             const int* __restrict__ batch,
                                             int* __restrict__ counts,
                                             int* __restrict__ lists) {
  int e = blockIdx.x * blockDim.x + threadIdx.x;
  if (e >= NEVENTS) return;
  int neuron = (int)events[2 * e + 1];
  int row = batch[e] * N_NEURONS + neuron;
  int slot = atomicAdd(&counts[row], 1);
  if (slot < MAXE) lists[row * MAXE + slot] = e;
}

// ---------------- kernel 2: one wave32 per row; WMMA tile reduction -------
__global__ __launch_bounds__(32) void k_accum(const float* __restrict__ events,
                                              const int* __restrict__ counts,
                                              const int* __restrict__ lists,
                                              float* __restrict__ out) {
  const int row  = blockIdx.x;
  const int lane = threadIdx.x;             // wave32: 0..31

  __shared__ int      s_idx[MAXE];
  __shared__ float    s_tau[MAXE];
  __shared__ unsigned s_mask[N_TILES / 32]; // 128-bit active-tile mask

  int c = counts[row];
  c = (c > MAXE) ? MAXE : c;

  // --- stage this row's event-index list into LDS (async path on CDNA5) ---
#if __has_builtin(__builtin_amdgcn_global_load_async_to_lds_b32)
  {
    const int* gsrc = lists + row * MAXE + lane;
    __builtin_amdgcn_global_load_async_to_lds_b32(
        (__attribute__((address_space(1))) int*)(uintptr_t)gsrc,
        (__attribute__((address_space(3))) int*)(uintptr_t)&s_idx[lane], 0, 0);
    __builtin_amdgcn_global_load_async_to_lds_b32(
        (__attribute__((address_space(1))) int*)(uintptr_t)(gsrc + 32),
        (__attribute__((address_space(3))) int*)(uintptr_t)&s_idx[lane + 32], 0, 0);
#if __has_builtin(__builtin_amdgcn_s_wait_asynccnt)
    __builtin_amdgcn_s_wait_asynccnt(0);
#else
    asm volatile("s_wait_asynccnt 0" ::: "memory");
#endif
  }
  __syncthreads();
  for (int j = lane; j < MAXE; j += 32)
    if (j >= c) s_idx[j] = 0x7fffffff;      // pad unused slots for the sort
#else
  for (int j = lane; j < MAXE; j += 32)
    s_idx[j] = (j < c) ? lists[row * MAXE + j] : 0x7fffffff;
#endif
  __syncthreads();

  // --- sort indices (odd-even) so the float sum order is deterministic ---
  for (int p = 0; p < MAXE; ++p) {
    int j = (p & 1) + 2 * lane;             // covers all adjacent pairs
    if (j + 1 < MAXE) {
      int a = s_idx[j], b = s_idx[j + 1];
      if (b < a) { s_idx[j] = b; s_idx[j + 1] = a; }
    }
    __syncthreads();
  }

  // --- gather event times; pad with -1e9 so padded slots contribute 0 ---
  for (int j = lane; j < MAXE; j += 32)
    s_tau[j] = (j < c) ? events[2 * s_idx[j]] : -1.0e9f;
  if (lane < N_TILES / 32) s_mask[lane] = 0u;
  __syncthreads();

  // --- active-tile mask: each event touches ~5 tiles (±32 samples) ---
  for (int j = lane; j < c; j += 32) {
    int ft = (int)(s_tau[j] * 0.0625f);     // tau/16, tau >= 0
    int lo = ft - 2; if (lo < 0) lo = 0;
    int hi = ft + 2; if (hi > N_TILES - 1) hi = N_TILES - 1;
    for (int tt = lo; tt <= hi; ++tt)
      atomicOr(&s_mask[tt >> 5], 1u << (tt & 31));
  }
  __syncthreads();

  // --- WMMA setup: A = ones(16x32, f16) -> D[m,n] = sum_k G[k,n] (+C f32)
  v16h ones;
#pragma unroll
  for (int j = 0; j < 16; ++j) ones[j] = (_Float16)1.0f;

  const int col   = lane & 15;              // output column (time within tile)
  const int kbase = (lane >> 4) << 4;       // K half handled by this lane
  const int ngroups = (c > 32) ? 2 : 1;     // uniform branch
  float* orow = out + (size_t)row * SEQ_LEN;

  for (int tt = 0; tt < N_TILES; ++tt) {
    if (!((s_mask[tt >> 5] >> (tt & 31)) & 1u)) continue;  // uniform skip
    const float tf = (float)(tt * 16 + col);
    v8f acc = {};
    for (int g = 0; g < ngroups; ++g) {
      v16h bg;                              // G tile: 32 events x 16 times
#pragma unroll
      for (int j = 0; j < 16; ++j) {
        float d   = tf - s_tau[g * 32 + kbase + j];
        float val = COEF * fast_exp2(d * d * QSCALE);  // 0 when out of window
        bg[j] = (_Float16)val;
      }
      // D = ones x G + acc   (matrix pipe does the event reduction)
      acc = __builtin_amdgcn_wmma_f32_16x16x32_f16(
          /*neg_a=*/false, ones, /*neg_b=*/false, bg,
          /*c_mod=*/(short)0, acc, /*reuse_a=*/false, /*reuse_b=*/false);
    }
    // Row 0 of D lives in VGPR0 of lanes 0..15 (documented C/D layout)
    if (lane < 16) orow[tt * 16 + col] = acc[0];
  }
}

// --------------------------------------------------------------------------
extern "C" void kernel_launch(void* const* d_in, const int* in_sizes, int n_in,
                              void* d_out, int out_size, void* d_ws, size_t ws_size,
                              hipStream_t stream) {
  const float* events = (const float*)d_in[0];  // (65536, 2) [time, neuron]
  const int*   batch  = (const int*)d_in[1];    // (65536,)
  float*       out    = (float*)d_out;          // (16, 512, 2048)

  int* counts = (int*)d_ws;                     // ROWS ints
  int* lists  = counts + ROWS;                  // ROWS*MAXE ints (~2 MB total)

  k_zero<<<(OUT_ELEMS / 4 + 255) / 256, 256, 0, stream>>>((float4*)out, counts);
  k_bin<<<(NEVENTS + 255) / 256, 256, 0, stream>>>(events, batch, counts, lists);
  k_accum<<<ROWS, 32, 0, stream>>>(events, counts, lists, out);
}
